// CrossAttention_28810640621651
// MI455X (gfx1250) — matrix operand order repaired and checked by reading, not compiled, not run on hardware
//
#include <hip/hip_runtime.h>

#define S_LEN  4096
#define DMODEL 320
#define NH     8
#define DHEAD  40
#define DPAD   64   // QK K-dim padded (zero pad -> exact)
#define DVPAD  48   // AV N-dim padded to 3x16
#define QSCALE 0.15811388300841897f  // 1/sqrt(40)

typedef _Float16 half8_t __attribute__((ext_vector_type(8)));
typedef _Float16 v16h    __attribute__((ext_vector_type(16)));
typedef float    v8f     __attribute__((ext_vector_type(8)));

// A-fragment loader (16x32 f16 A, this file's contiguous k order, B loads match):
// lanes0-15: K {0..15}; lanes16-31: K {16..31}; row = lane&15.
__device__ __forceinline__ v16h load_a_frag(const _Float16* rowbase_kb, int lh) {
  half8_t lo = *(const half8_t*)(rowbase_kb + lh * 16);
  half8_t hi = *(const half8_t*)(rowbase_kb + lh * 16 + 8);
  v16h a;
#pragma unroll
  for (int e = 0; e < 8; ++e) { a[e] = lo[e]; a[e + 8] = hi[e]; }
  return a;
}

// ---- VALU cross-lane reductions within 16-lane rows via v_permlane16_b32 ---
__device__ __forceinline__ float permlane16_f(float v, unsigned s0, unsigned s1) {
  int x = __float_as_int(v);
  return __int_as_float(__builtin_amdgcn_permlane16(x, x, s0, s1, false, false));
}
__device__ __forceinline__ float rowmax16(float v) {
  v = fmaxf(v, permlane16_f(v, 0x67452301u, 0xEFCDAB89u));  // xor 1
  v = fmaxf(v, permlane16_f(v, 0x45670123u, 0xCDEF89ABu));  // xor 2
  v = fmaxf(v, permlane16_f(v, 0x01234567u, 0x89ABCDEFu));  // xor 4
  v = fmaxf(v, permlane16_f(v, 0xFEDCBA98u, 0x76543210u));  // xor 8
  return v;
}
__device__ __forceinline__ float rowsum16(float v) {
  v += permlane16_f(v, 0x67452301u, 0xEFCDAB89u);
  v += permlane16_f(v, 0x45670123u, 0xCDEF89ABu);
  v += permlane16_f(v, 0x01234567u, 0x89ABCDEFu);
  v += permlane16_f(v, 0xFEDCBA98u, 0x76543210u);
  return v;
}

// ---------------- fp32 -> f16 convert --------------------------------------
__global__ void cvt_f16_kernel(const float* __restrict__ src,
                               _Float16* __restrict__ dst, int n) {
  int i = blockIdx.x * blockDim.x + threadIdx.x;
  if (i < n) dst[i] = (_Float16)src[i];
}

// ---------------- f16 zero fill (for padded buffers) ------------------------
__global__ void zero_f16_kernel(_Float16* __restrict__ dst, int n) {
  int i = blockIdx.x * blockDim.x + threadIdx.x;
  if (i < n) dst[i] = (_Float16)0.0f;
}

// ---------------- QKV projection: one wave per 16x16 output tile ------------
// q = (xh @ Wq^T)*scale -> qp[h][s][DPAD]; k -> kp[h][s][DPAD]; v -> vt[h][d][s]
__global__ __launch_bounds__(32) void qkv_gemm_kernel(
    const _Float16* __restrict__ xh,
    const _Float16* __restrict__ wqh, const _Float16* __restrict__ wkh,
    const _Float16* __restrict__ wvh,
    _Float16* __restrict__ qp, _Float16* __restrict__ kp,
    _Float16* __restrict__ vt) {
  const int lane = threadIdx.x, lh = lane >> 4, ln = lane & 15;
  const int ot = blockIdx.x % (DMODEL / 16);
  const int st = blockIdx.x / (DMODEL / 16);
  const int s0 = st * 16, o0 = ot * 16;

  const _Float16* arow = xh + (size_t)(s0 + ln) * DMODEL;
  const _Float16* bq   = wqh + (size_t)(o0 + ln) * DMODEL;
  const _Float16* bk   = wkh + (size_t)(o0 + ln) * DMODEL;
  const _Float16* bv   = wvh + (size_t)(o0 + ln) * DMODEL;

  v8f cq = {}, ck = {}, cv = {};
#pragma unroll
  for (int kb = 0; kb < DMODEL; kb += 32) {
    v16h a  = load_a_frag(arow + kb, lh);
    v16h b0 = *(const v16h*)(bq + kb + lh * 16);
    v16h b1 = *(const v16h*)(bk + kb + lh * 16);
    v16h b2 = *(const v16h*)(bv + kb + lh * 16);
    cq = __builtin_amdgcn_wmma_f32_16x16x32_f16(false, a, false, b0, (short)0, cq, false, false);
    ck = __builtin_amdgcn_wmma_f32_16x16x32_f16(false, a, false, b1, (short)0, ck, false, false);
    cv = __builtin_amdgcn_wmma_f32_16x16x32_f16(false, a, false, b2, (short)0, cv, false, false);
  }
  // epilogue: C layout lane = col N -> o, VGPR r -> row M = r + 8*lh -> s
  const int o = o0 + ln;
  const int h = o / DHEAD, d = o % DHEAD;
#pragma unroll
  for (int r = 0; r < 8; ++r) {
    int s = s0 + r + 8 * lh;
    qp[((size_t)h * S_LEN + s) * DPAD + d] = (_Float16)(cq[r] * QSCALE);
    kp[((size_t)h * S_LEN + s) * DPAD + d] = (_Float16)ck[r];
    vt[((size_t)h * DVPAD + d) * S_LEN + s] = (_Float16)cv[r];
  }
}

// ---------------- Flash attention: one wave per (head, 16-row tile) ---------
// 64-column j-chunks: 8 QK WMMAs + 6 AV WMMAs per chunk. ctx written as f16.
__global__ __launch_bounds__(32) void attn_kernel(const _Float16* __restrict__ qp,
                                                  const _Float16* __restrict__ kp,
                                                  const _Float16* __restrict__ vt,
                                                  _Float16* __restrict__ ctxh) {
  __shared__ __align__(32) _Float16 pbuf[16][64];
  const int lane = threadIdx.x, lh = lane >> 4, ln = lane & 15;
  const int it = blockIdx.x & (S_LEN / 16 - 1);
  const int h  = blockIdx.x >> 8;
  const int i0 = it * 16;

  // Q as two A-fragments over padded K=64
  const _Float16* qb = qp + ((size_t)h * S_LEN + i0 + ln) * DPAD;
  v16h aq0 = load_a_frag(qb, lh);
  v16h aq1 = load_a_frag(qb + 32, lh);

  const _Float16* kbase = kp + (size_t)h * S_LEN * DPAD;
  const _Float16* vbase = vt + (size_t)h * DVPAD * S_LEN;

  v8f acc[3] = {{}, {}, {}};
  float m[8], l[8];
#pragma unroll
  for (int r = 0; r < 8; ++r) { m[r] = -__builtin_inff(); l[r] = 0.0f; }

  for (int jc = 0; jc < S_LEN / 64; ++jc) {
    const int j0 = jc * 64;

    // prefetch next chunk's K rows and V row-segments into cache
    if (jc + 1 < S_LEN / 64) {
      const int jn = j0 + 64;
      __builtin_prefetch(kbase + (size_t)(jn + lane) * DPAD, 0, 3);
      __builtin_prefetch(kbase + (size_t)(jn + 32 + lane) * DPAD, 0, 3);
      __builtin_prefetch(vbase + (size_t)lane * S_LEN + jn, 0, 3);
      __builtin_prefetch(vbase + (size_t)(16 + lane) * S_LEN + jn, 0, 3);
    }

    v8f c[4];
    // ---- S = Q K^T over 4 subtiles of 16 keys ---------------------------
#pragma unroll
    for (int sub = 0; sub < 4; ++sub) {
      const _Float16* kb = kbase + (size_t)(j0 + sub * 16 + ln) * DPAD;
      v16h b0 = *(const v16h*)(kb + lh * 16);
      v16h b1 = *(const v16h*)(kb + 32 + lh * 16);
      v8f cc = {};
      cc = __builtin_amdgcn_wmma_f32_16x16x32_f16(false, aq0, false, b0, (short)0, cc, false, false);
      cc = __builtin_amdgcn_wmma_f32_16x16x32_f16(false, aq1, false, b1, (short)0, cc, false, false);
      c[sub] = cc;
    }
    // ---- online softmax over this 64-col chunk --------------------------
#pragma unroll
    for (int r = 0; r < 8; ++r) {
      float cm = fmaxf(fmaxf(c[0][r], c[1][r]), fmaxf(c[2][r], c[3][r]));
      cm = rowmax16(cm);
      float mn = fmaxf(m[r], cm);
      float sc = __expf(m[r] - mn);
      m[r] = mn;
      float p0 = __expf(c[0][r] - mn);
      float p1 = __expf(c[1][r] - mn);
      float p2 = __expf(c[2][r] - mn);
      float p3 = __expf(c[3][r] - mn);
      l[r] = l[r] * sc + (p0 + p1) + (p2 + p3);
      acc[0][r] *= sc; acc[1][r] *= sc; acc[2][r] *= sc;
      int M = r + 8 * lh;
      pbuf[M][ln]      = (_Float16)p0;
      pbuf[M][16 + ln] = (_Float16)p1;
      pbuf[M][32 + ln] = (_Float16)p2;
      pbuf[M][48 + ln] = (_Float16)p3;
    }
    __syncthreads();
    // ---- P (16x64) as two A-fragments -----------------------------------
    const _Float16* pr = &pbuf[ln][0];
    v16h ap0 = load_a_frag(pr, lh);
    v16h ap1 = load_a_frag(pr + 32, lh);
    // ---- O += P @ V : 3 N-tiles x 2 K-steps ------------------------------
#pragma unroll
    for (int n = 0; n < 3; ++n) {
      const _Float16* vb = vbase + (size_t)(n * 16 + ln) * S_LEN + j0;
      v16h b0 = *(const v16h*)(vb + lh * 16);
      v16h b1 = *(const v16h*)(vb + 32 + lh * 16);
      acc[n] = __builtin_amdgcn_wmma_f32_16x16x32_f16(false, ap0, false, b0, (short)0, acc[n], false, false);
      acc[n] = __builtin_amdgcn_wmma_f32_16x16x32_f16(false, ap1, false, b1, (short)0, acc[n], false, false);
    }
    __syncthreads();
  }

  // ---- finalize: divide by row sums, write ctxh[s][h*DH + d] as f16 ------
#pragma unroll
  for (int r = 0; r < 8; ++r) {
    float s = rowsum16(l[r]);
    float rinv = 1.0f / s;
    int row = i0 + r + 8 * lh;
    _Float16* cr = ctxh + (size_t)row * DMODEL + h * DHEAD;
    cr[ln]      = (_Float16)(acc[0][r] * rinv);
    cr[16 + ln] = (_Float16)(acc[1][r] * rinv);
    if (ln < 8) cr[32 + ln] = (_Float16)(acc[2][r] * rinv);  // d = 32..39
  }
}

// ---------------- output projection: out = ctxh @ Wo^T + bo (WMMA) ----------
__global__ __launch_bounds__(32) void o_gemm_kernel(const _Float16* __restrict__ ctxh,
                                                    const _Float16* __restrict__ woh,
                                                    const float* __restrict__ bo,
                                                    float* __restrict__ out) {
  const int lane = threadIdx.x, lh = lane >> 4, ln = lane & 15;
  const int ot = blockIdx.x % (DMODEL / 16);
  const int st = blockIdx.x / (DMODEL / 16);
  const int s0 = st * 16, o0 = ot * 16;

  const _Float16* arow = ctxh + (size_t)(s0 + ln) * DMODEL;
  const _Float16* brow = woh + (size_t)(o0 + ln) * DMODEL;
  v8f cacc = {};
#pragma unroll
  for (int kb = 0; kb < DMODEL; kb += 32) {
    v16h a = load_a_frag(arow + kb, lh);
    v16h b = *(const v16h*)(brow + kb + lh * 16);
    cacc = __builtin_amdgcn_wmma_f32_16x16x32_f16(false, a, false, b, (short)0, cacc, false, false);
  }
  const int o = o0 + ln;
  const float bias = bo[o];
#pragma unroll
  for (int r = 0; r < 8; ++r) {
    int s = s0 + r + 8 * lh;
    out[(size_t)s * DMODEL + o] = cacc[r] + bias;
  }
}

extern "C" void kernel_launch(void* const* d_in, const int* in_sizes, int n_in,
                              void* d_out, int out_size, void* d_ws, size_t ws_size,
                              hipStream_t stream) {
  (void)in_sizes; (void)n_in; (void)out_size; (void)ws_size;
  const float* x  = (const float*)d_in[0];
  const float* Wq = (const float*)d_in[1];
  const float* Wk = (const float*)d_in[2];
  const float* Wv = (const float*)d_in[3];
  const float* Wo = (const float*)d_in[4];
  const float* bo = (const float*)d_in[5];
  float* out = (float*)d_out;

  char* ws = (char*)d_ws;
  size_t off = 0;
  _Float16* xh  = (_Float16*)(ws + off); off += (size_t)S_LEN * DMODEL * 2;
  _Float16* wqh = (_Float16*)(ws + off); off += (size_t)DMODEL * DMODEL * 2;
  _Float16* wkh = (_Float16*)(ws + off); off += (size_t)DMODEL * DMODEL * 2;
  _Float16* wvh = (_Float16*)(ws + off); off += (size_t)DMODEL * DMODEL * 2;
  _Float16* woh = (_Float16*)(ws + off); off += (size_t)DMODEL * DMODEL * 2;
  _Float16* qp  = (_Float16*)(ws + off); off += (size_t)NH * S_LEN * DPAD * 2;
  _Float16* kp  = (_Float16*)(ws + off); off += (size_t)NH * S_LEN * DPAD * 2;
  _Float16* vt  = (_Float16*)(ws + off); off += (size_t)NH * DVPAD * S_LEN * 2;
  _Float16* ctxh = (_Float16*)(ws + off);

  const int W2 = DMODEL * DMODEL;
  cvt_f16_kernel<<<(S_LEN * DMODEL + 255) / 256, 256, 0, stream>>>(x, xh, S_LEN * DMODEL);
  cvt_f16_kernel<<<(W2 + 255) / 256, 256, 0, stream>>>(Wq, wqh, W2);
  cvt_f16_kernel<<<(W2 + 255) / 256, 256, 0, stream>>>(Wk, wkh, W2);
  cvt_f16_kernel<<<(W2 + 255) / 256, 256, 0, stream>>>(Wv, wvh, W2);
  cvt_f16_kernel<<<(W2 + 255) / 256, 256, 0, stream>>>(Wo, woh, W2);

  {  // qp, kp, vt are contiguous in ws -> single zero fill
    int nz = NH * S_LEN * DPAD * 2 + NH * DVPAD * S_LEN;
    zero_f16_kernel<<<(nz + 255) / 256, 256, 0, stream>>>(qp, nz);
  }

  qkv_gemm_kernel<<<(S_LEN / 16) * (DMODEL / 16), 32, 0, stream>>>(
      xh, wqh, wkh, wvh, qp, kp, vt);
  attn_kernel<<<NH * (S_LEN / 16), 32, 0, stream>>>(qp, kp, vt, ctxh);
  o_gemm_kernel<<<(S_LEN / 16) * (DMODEL / 16), 32, 0, stream>>>(ctxh, woh, bo, out);
}
